// BilateralGPT_57526791962858
// MI455X (gfx1250) — matrix operand order repaired and checked by reading, not compiled, not run on hardware
//
#include <hip/hip_runtime.h>
#include <hip/hip_bf16.h>
#include <math.h>

// ---------------------------------------------------------------------------
// BilateralGPT forward for gfx1250 (MI455X)
//   - WMMA f32_16x16x32_f16 GEMM core, double-buffered LDS
//   - A operand kept in f16 (producers emit f16), staged via async-to-LDS
// ---------------------------------------------------------------------------

typedef __attribute__((ext_vector_type(16))) _Float16 v16h;
typedef __attribute__((ext_vector_type(8)))  float    v8f;

typedef int v4i __attribute__((vector_size(16)));
typedef __attribute__((address_space(1))) v4i global_v4i;   // global
typedef __attribute__((address_space(3))) v4i lds_v4i;      // LDS

static constexpr int Vsz = 32000, D = 768, NH = 12, DHd = 64, NLs = 6,
                     LATd = 128, NLAYERn = 6, Bbb = 2, Ttt = 1024;
static constexpr int NT  = Bbb * Ttt;   // 2048 rows
static constexpr float EPSf = 1e-5f;

#define BM 128
#define BN 128
#define BK 32

#if __has_builtin(__builtin_amdgcn_global_load_async_to_lds_b128)
#define ASYNC_A 1
#else
#define ASYNC_A 0
#endif

__device__ __forceinline__ void wait_async0()
{
#if ASYNC_A
#if __has_builtin(__builtin_amdgcn_s_wait_asynccnt)
    __builtin_amdgcn_s_wait_asynccnt(0);
#else
    asm volatile("s_wait_asynccnt 0x0" ::: "memory");
#endif
#endif
}

__device__ __forceinline__ unsigned pack_f16x2(float a, float b)
{
    union { _Float16 h[2]; unsigned u; } p;
    p.h[0] = (_Float16)a;
    p.h[1] = (_Float16)b;
    return p.u;
}

// ------------------------------ WMMA GEMM ----------------------------------
// C[M,N] = A[M,K] @ B[K,N] (+C if addC). A f16 row-major, B/C fp32 row-major.
// Block: 256 threads = 8 waves arranged 4(m) x 2(n); wave -> 32x64 sub-tile.
__global__ __launch_bounds__(256)
void gemm_wmma_f16(const _Float16* __restrict__ A, const float* __restrict__ Bm,
                   float* __restrict__ C, int M, int N, int K,
                   int lda, int ldb, int ldc, int addC)
{
    __shared__ _Float16 As[2][BM][BK];   // row-major  [m][k]
    __shared__ _Float16 Bs[2][BN][BK];   // transposed [n][k]

    const int tid  = threadIdx.x;
    const int lane = tid & 31;
    const int wid  = tid >> 5;
    const int wm   = wid & 3;            // 0..3 -> 32-row strip
    const int wn   = wid >> 2;           // 0..1 -> 64-col strip
    const int bn0  = blockIdx.x * BN;
    const int bm0  = blockIdx.y * BM;

    v8f acc[2][4];
#pragma unroll
    for (int i = 0; i < 2; ++i)
#pragma unroll
        for (int j = 0; j < 4; ++j) acc[i][j] = {};

    // ---- A tile staging: 512 x 16B chunks, 2 per thread -------------------
    auto stageA = [&](int nb, int k0) {
#pragma unroll
        for (int i = 0; i < 2; ++i) {
            const int chunk = tid + i * 256;          // 0..511
            const int r  = chunk >> 2;                // row 0..127
            const int cb = (chunk & 3) * 8;           // f16 elem offset
            const _Float16* g = A + (size_t)(bm0 + r) * lda + k0 + cb;
#if ASYNC_A
            __builtin_amdgcn_global_load_async_to_lds_b128(
                (global_v4i*)g, (lds_v4i*)&As[nb][r][cb], 0, 0);
#else
            *(uint4*)&As[nb][r][cb] = *(const uint4*)g;
#endif
        }
    };

    // ---- B tile staging: each thread owns two 2(k)x4(n) chunks ------------
    // task = tid + i*256 : kp = task>>5 (k-pair 0..15), cg = task&31 (col grp)
    float4 breg[2][2];                    // [task][k row 0/1]
    auto loadB = [&](int k0) {
#pragma unroll
        for (int i = 0; i < 2; ++i) {
            const int task = tid + i * 256;
            const int kp = task >> 5;
            const int cg = task & 31;
            const float* p0 = Bm + (size_t)(k0 + 2 * kp) * ldb + bn0 + cg * 4;
            breg[i][0] = *(const float4*)(p0);
            breg[i][1] = *(const float4*)(p0 + ldb);
        }
        if (k0 + BK < K)   // L2 hint one tile further ahead
            __builtin_prefetch(Bm + (size_t)(k0 + BK + (tid >> 3)) * ldb + bn0 + (tid & 7) * 16, 0, 0);
    };
    auto storeB = [&](int nb) {
#pragma unroll
        for (int i = 0; i < 2; ++i) {
            const int task = tid + i * 256;
            const int kp = task >> 5;
            const int cg = task & 31;
            const float4 f0 = breg[i][0], f1 = breg[i][1];
            *(unsigned*)&Bs[nb][cg * 4 + 0][2 * kp] = pack_f16x2(f0.x, f1.x);
            *(unsigned*)&Bs[nb][cg * 4 + 1][2 * kp] = pack_f16x2(f0.y, f1.y);
            *(unsigned*)&Bs[nb][cg * 4 + 2][2 * kp] = pack_f16x2(f0.z, f1.z);
            *(unsigned*)&Bs[nb][cg * 4 + 3][2 * kp] = pack_f16x2(f0.w, f1.w);
        }
    };

    // ---- fragment loads (all up front) + 8 WMMAs --------------------------
    auto compute = [&](int nb) {
        const int half = lane >> 4;
        v16h af[2], bf[4];
        // A 16-bit 16x32: lane<16 K{0..7,16..23}, lane>=16 K{8..15,24..31}
#pragma unroll
        for (int mi = 0; mi < 2; ++mi) {
            const _Float16* rp = &As[nb][wm * 32 + mi * 16 + (lane & 15)][0];
            union { uint4 u[2]; v16h v; } fu;
            fu.u[0] = *(const uint4*)(rp + half * 8);
            fu.u[1] = *(const uint4*)(rp + half * 8 + 16);
            af[mi] = fu.v;
        }
        // B 16-bit 32x16: lane<16 K{0..7,16..23}, lane>=16 K{8..15,24..31}
#pragma unroll
        for (int ni = 0; ni < 4; ++ni) {
            const _Float16* cp = &Bs[nb][wn * 64 + ni * 16 + (lane & 15)][0];
            union { uint4 u[2]; v16h v; } fu;
            fu.u[0] = *(const uint4*)(cp + half * 8);
            fu.u[1] = *(const uint4*)(cp + half * 8 + 16);
            bf[ni] = fu.v;
        }
#pragma unroll
        for (int ni = 0; ni < 4; ++ni)
#pragma unroll
            for (int mi = 0; mi < 2; ++mi)
                acc[mi][ni] = __builtin_amdgcn_wmma_f32_16x16x32_f16(
                    false, af[mi], false, bf[ni], (short)0, acc[mi][ni], false, false);
    };

    // ---- software-pipelined main loop (double buffer) ---------------------
    stageA(0, 0);
    loadB(0);
    storeB(0);
    wait_async0();
    __syncthreads();

    int cur = 0;
    for (int k0 = 0; k0 < K; k0 += BK) {
        const int  k1  = k0 + BK;
        const bool nxt = k1 < K;
        if (nxt) { stageA(cur ^ 1, k1); loadB(k1); }   // overlap with WMMA
        compute(cur);
        if (nxt) { storeB(cur ^ 1); }
        wait_async0();
        __syncthreads();
        cur ^= 1;
    }

    // ---- epilogue: C/D layout -> lane<16: M=r, lane>=16: M=8+r; N=lane&15 -
    const int moff = (lane >> 4) * 8;
#pragma unroll
    for (int mi = 0; mi < 2; ++mi)
#pragma unroll
        for (int ni = 0; ni < 4; ++ni) {
            const int col = bn0 + wn * 64 + ni * 16 + (lane & 15);
#pragma unroll
            for (int r = 0; r < 8; ++r) {
                const int row = bm0 + wm * 32 + mi * 16 + moff + r;
                const size_t o = (size_t)row * ldc + col;
                float v = acc[mi][ni][r];
                if (addC) v += C[o];
                C[o] = v;
            }
        }
}

// ------------------------------ helpers ------------------------------------
__global__ void embed_k(const int* __restrict__ idx, const float* __restrict__ tok,
                        const float* __restrict__ pos, float* __restrict__ xm,
                        float* __restrict__ xa)
{
    const int row = blockIdx.x;
    const int t = row % Ttt;
    const int tk = idx[row];
    for (int d = threadIdx.x; d < D; d += 256) {
        float v = tok[(size_t)tk * D + d] + pos[(size_t)t * D + d];
        xm[(size_t)row * D + d] = v;
        xa[(size_t)row * D + d] = v;
    }
}

// LayerNorm: fp32 in, f16 out (feeds WMMA A operand directly)
__global__ __launch_bounds__(256)
void layernorm_k(const float* __restrict__ x, const float* __restrict__ g,
                 const float* __restrict__ b, _Float16* __restrict__ y)
{
    __shared__ float red[256];
    const int row = blockIdx.x, tid = threadIdx.x;
    const float* xr = x + (size_t)row * D;
    float s = 0.f;
    for (int d = tid; d < D; d += 256) s += xr[d];
    red[tid] = s; __syncthreads();
    for (int st = 128; st > 0; st >>= 1) { if (tid < st) red[tid] += red[tid + st]; __syncthreads(); }
    const float mean = red[0] / (float)D;
    __syncthreads();
    float v = 0.f;
    for (int d = tid; d < D; d += 256) { float t = xr[d] - mean; v += t * t; }
    red[tid] = v; __syncthreads();
    for (int st = 128; st > 0; st >>= 1) { if (tid < st) red[tid] += red[tid + st]; __syncthreads(); }
    const float rstd = rsqrtf(red[0] / (float)D + EPSf);
    _Float16* yr = y + (size_t)row * D;
    for (int d = tid; d < D; d += 256)
        yr[d] = (_Float16)((xr[d] - mean) * rstd * g[d] + b[d]);
}

// concat fp32 (a,b) -> f16 [row][2D]
__global__ void concat_k(const float* __restrict__ a, const float* __restrict__ b,
                         _Float16* __restrict__ c)
{
    const int row = blockIdx.x;
    for (int d = threadIdx.x; d < D; d += 256) {
        c[(size_t)row * 2 * D + d]     = (_Float16)a[(size_t)row * D + d];
        c[(size_t)row * 2 * D + D + d] = (_Float16)b[(size_t)row * D + d];
    }
}

__global__ void sigmul_k(float* __restrict__ x, const float* __restrict__ g, int n)
{
    const int i = blockIdx.x * 256 + threadIdx.x;
    if (i < n) x[i] *= 1.f / (1.f + expf(-g[i]));
}

// exact GELU, fp32 in -> f16 out (feeds next GEMM's A operand)
__global__ void gelu_k(const float* __restrict__ x, _Float16* __restrict__ y, int n)
{
    const int i = blockIdx.x * 256 + threadIdx.x;
    if (i < n) {
        float v = x[i];
        y[i] = (_Float16)(0.5f * v * (1.f + erff(v * 0.70710678118654752f)));
    }
}

__global__ void rope_k(float* __restrict__ x)   // [NT][NH][DHd], pairs (2i,2i+1)
{
    const int row = blockIdx.x;
    const int t = row % Ttt;
    const int i = threadIdx.x;                  // 0..383 (NH*DHd/2)
    if (i < NH * DHd / 2) {
        const int h = i >> 5, p = i & 31;
        const float freq = expf(-(float)(2 * p) * (9.210340371976184f / (float)DHd));
        const float ang = (float)t * freq;
        float sn, cs; sincosf(ang, &sn, &cs);
        const size_t o = (size_t)row * D + h * DHd + 2 * p;
        const float x0 = x[o], x1 = x[o + 1];
        x[o]     = x0 * cs - x1 * sn;
        x[o + 1] = x0 * sn + x1 * cs;
    }
}

__global__ void lk_gemm_k(const float* __restrict__ lk, const float* __restrict__ Wlk,
                          float* __restrict__ out)    // [NLs][D]
{
    const int i = blockIdx.x * 256 + threadIdx.x;
    if (i < NLs * D) {
        const int r = i / D, c = i % D;
        float a = 0.f;
        for (int l = 0; l < LATd; ++l) a += lk[r * LATd + l] * Wlk[(size_t)l * D + c];
        out[i] = a;
    }
}

// One block per (b,h,t) query row; scores in LDS, two-pass softmax, then AV.
__global__ __launch_bounds__(128)
void attn_k(const float* __restrict__ q, const float* __restrict__ k,
            const float* __restrict__ v, const float* __restrict__ lkb,
            _Float16* __restrict__ out)
{
    __shared__ float sc[Ttt + NLs];
    __shared__ float qs[DHd];
    __shared__ float red[128];
    const int bid = blockIdx.x;
    const int t = bid % Ttt;
    const int h = (bid / Ttt) % NH;
    const int b = bid / (Ttt * NH);
    const int tid = threadIdx.x;
    const size_t qo = (size_t)(b * Ttt + t) * D + h * DHd;

    if (tid < DHd) qs[tid] = q[qo + tid];
    __syncthreads();

    const int cnt = t + 1 + NLs;                 // causal keys + latent slots
    for (int j = tid; j < cnt; j += 128) {
        const float* kr = (j <= t) ? (k + (size_t)(b * Ttt + j) * D + h * DHd)
                                   : (lkb + (size_t)(j - (t + 1)) * D + h * DHd);
        float s = 0.f;
#pragma unroll
        for (int d = 0; d < DHd; ++d) s += qs[d] * kr[d];
        sc[j] = s * 0.125f;                      // 1/sqrt(64)
    }
    __syncthreads();

    float mx = -1e30f;
    for (int j = tid; j < cnt; j += 128) mx = fmaxf(mx, sc[j]);
    red[tid] = mx; __syncthreads();
    for (int st = 64; st > 0; st >>= 1) { if (tid < st) red[tid] = fmaxf(red[tid], red[tid + st]); __syncthreads(); }
    mx = red[0]; __syncthreads();

    float sm = 0.f;
    for (int j = tid; j < cnt; j += 128) { float e = expf(sc[j] - mx); sc[j] = e; sm += e; }
    red[tid] = sm; __syncthreads();
    for (int st = 64; st > 0; st >>= 1) { if (tid < st) red[tid] += red[tid + st]; __syncthreads(); }
    const float inv = 1.f / red[0];
    __syncthreads();

    if (tid < DHd) {                             // latent V rows are zero -> skip
        float a = 0.f;
        for (int j = 0; j <= t; ++j)
            a += sc[j] * v[(size_t)(b * Ttt + j) * D + h * DHd + tid];
        out[qo + tid] = (_Float16)(a * inv);
    }
}

// ------------------------------ host side ----------------------------------
extern "C" void kernel_launch(void* const* d_in, const int* in_sizes, int n_in,
                              void* d_out, int out_size, void* d_ws, size_t ws_size,
                              hipStream_t stream)
{
    (void)in_sizes; (void)n_in; (void)out_size; (void)ws_size;

    // jax tree_leaves (sorted-key) order of setup_inputs():
    // 0: idx; per block (22 leaves): attn{Wk,Wlk,Wlq,Wproj,Wq,Wv,lk,lq},
    // ff_W1, ff_W2, lat_post{gate,ln_b,ln_g,proj}, lat_pre{gate,ln_b,ln_g,proj},
    // ln1_b, ln1_g, ln2_b, ln2_g; then head_w, ln_f_b, ln_f_g, pos_emb, tok_emb.
    const int* idx = (const int*)d_in[0];
    auto BLKP = [&](int L, int o) -> const float* { return (const float*)d_in[1 + L * 22 + o]; };
    const float* head_w  = (const float*)d_in[1 + NLAYERn * 22 + 0];
    const float* ln_f_b  = (const float*)d_in[1 + NLAYERn * 22 + 1];
    const float* ln_f_g  = (const float*)d_in[1 + NLAYERn * 22 + 2];
    const float* pos_emb = (const float*)d_in[1 + NLAYERn * 22 + 3];
    const float* tok_emb = (const float*)d_in[1 + NLAYERn * 22 + 4];

    float* ws = (float*)d_ws;
    const size_t U = (size_t)NT * D;   // 1.57M floats
    float*     xm   = ws;                      // f32
    float*     xa   = ws + U;                  // f32
    _Float16*  tlnh = (_Float16*)(ws + 2 * U); // NT*D f16 (0.5U)
    float*     lm   = ws + 3 * U;
    float*     la   = ws + 4 * U;
    _Float16*  c0h  = (_Float16*)(ws + 5 * U); // NT*2D f16 (1U)
    _Float16*  c1h  = (_Float16*)(ws + 6 * U); // NT*2D f16 (1U)
    float*     g0   = ws + 7 * U;
    float*     g1   = ws + 8 * U;
    float*     qb   = ws + 9 * U;
    float*     kb   = ws + 10 * U;
    float*     vb   = ws + 11 * U;
    _Float16*  atth = (_Float16*)(ws + 12 * U);// NT*D f16 (0.5U)
    float*     ff1  = ws + 13 * U;             // NT*4D f32 (4U)
    _Float16*  ff1h = (_Float16*)(ws + 17 * U);// NT*4D f16 (2U)
    float*     lkb  = ws + 19 * U;             // NLs*D

    auto gemm = [&](const _Float16* A, const float* Bm, float* C,
                    int M, int N, int K, int lda, int ldb, int ldc, int addC) {
        dim3 grid(N / BN, M / BM);
        gemm_wmma_f16<<<grid, dim3(256), 0, stream>>>(A, Bm, C, M, N, K, lda, ldb, ldc, addC);
    };

    embed_k<<<NT, 256, 0, stream>>>(idx, tok_emb, pos_emb, xm, xa);

    for (int L = 0; L < NLAYERn; ++L) {
        const float* Wk    = BLKP(L, 0);
        const float* Wlk   = BLKP(L, 1);
        const float* Wproj = BLKP(L, 3);
        const float* Wq    = BLKP(L, 4);
        const float* Wv    = BLKP(L, 5);
        const float* lkw   = BLKP(L, 6);
        const float* ffW1  = BLKP(L, 8);
        const float* ffW2  = BLKP(L, 9);
        const float* po_gate = BLKP(L, 10); const float* po_lnb = BLKP(L, 11);
        const float* po_lng  = BLKP(L, 12); const float* po_proj = BLKP(L, 13);
        const float* pr_gate = BLKP(L, 14); const float* pr_lnb = BLKP(L, 15);
        const float* pr_lng  = BLKP(L, 16); const float* pr_proj = BLKP(L, 17);
        const float* ln1b = BLKP(L, 18); const float* ln1g = BLKP(L, 19);
        const float* ln2b = BLKP(L, 20); const float* ln2g = BLKP(L, 21);

        auto lateral = [&](const float* gate, const float* lnb, const float* lng,
                           const float* proj) {
            layernorm_k<<<NT, 256, 0, stream>>>(xm, lng, lnb, tlnh);
            gemm(tlnh, proj, lm, NT, D, D, D, D, D, 0);
            layernorm_k<<<NT, 256, 0, stream>>>(xa, lng + D, lnb + D, tlnh);
            gemm(tlnh, proj + (size_t)D * D, la, NT, D, D, D, D, D, 0);
            concat_k<<<NT, 256, 0, stream>>>(xm, la, c0h);
            concat_k<<<NT, 256, 0, stream>>>(xa, lm, c1h);
            gemm(c0h, gate, g0, NT, D, 2 * D, 2 * D, D, D, 0);
            gemm(c1h, gate + (size_t)2 * D * D, g1, NT, D, 2 * D, 2 * D, D, D, 0);
            const int n = NT * D;
            sigmul_k<<<(n + 255) / 256, 256, 0, stream>>>(xm, g0, n);
            sigmul_k<<<(n + 255) / 256, 256, 0, stream>>>(xa, g1, n);
        };

        auto attention = [&](float* x, int s) {
            layernorm_k<<<NT, 256, 0, stream>>>(x, ln1g + s * D, ln1b + s * D, tlnh);
            gemm(tlnh, Wq + (size_t)s * D * D, qb, NT, D, D, D, D, D, 0);
            gemm(tlnh, Wk + (size_t)s * D * D, kb, NT, D, D, D, D, D, 0);
            gemm(tlnh, Wv + (size_t)s * D * D, vb, NT, D, D, D, D, D, 0);
            rope_k<<<NT, 384, 0, stream>>>(qb);
            rope_k<<<NT, 384, 0, stream>>>(kb);
            lk_gemm_k<<<(NLs * D + 255) / 256, 256, 0, stream>>>(
                lkw + (size_t)s * NLs * LATd, Wlk + (size_t)s * LATd * D, lkb);
            attn_k<<<Bbb * NH * Ttt, 128, 0, stream>>>(qb, kb, vb, lkb, atth);
            gemm(atth, Wproj + (size_t)s * D * D, x, NT, D, D, D, D, D, 1);  // residual
        };

        auto ffn = [&](float* x, int s) {
            layernorm_k<<<NT, 256, 0, stream>>>(x, ln2g + s * D, ln2b + s * D, tlnh);
            gemm(tlnh, ffW1 + (size_t)s * D * 4 * D, ff1, NT, 4 * D, D, D, 4 * D, 4 * D, 0);
            const int n = NT * 4 * D;
            gelu_k<<<(n + 255) / 256, 256, 0, stream>>>(ff1, ff1h, n);
            gemm(ff1h, ffW2 + (size_t)s * 4 * D * D, x, NT, D, 4 * D, 4 * D, D, D, 1); // residual
        };

        lateral(pr_gate, pr_lnb, pr_lng, pr_proj);
        attention(xm, 0);
        attention(xa, 1);
        lateral(po_gate, po_lnb, po_lng, po_proj);
        ffn(xm, 0);
        ffn(xa, 1);
    }

    float* out = (float*)d_out;
    layernorm_k<<<NT, 256, 0, stream>>>(xm, ln_f_g, ln_f_b, tlnh);
    gemm(tlnh, head_w, out, NT, Vsz, D, D, Vsz, Vsz, 0);
    layernorm_k<<<NT, 256, 0, stream>>>(xa, ln_f_g + D, ln_f_b + D, tlnh);
    gemm(tlnh, head_w + (size_t)D * Vsz, out + (size_t)NT * Vsz, NT, Vsz, D, D, Vsz, Vsz, 0);
}
